// TemporalSelfAttention_43327630082523
// MI455X (gfx1250) — compile-verified, matrix operand order checked by reading, not run on hardware
//
#include <hip/hip_runtime.h>
#include <hip/hip_bf16.h>
#include <stdint.h>

// MI455X / gfx1250, wave32. GEMMs via v_wmma_f32_16x16x32_f16,
// all LDS staging via global_load_async_to_lds_b128 (ASYNCcnt).

typedef __attribute__((ext_vector_type(16))) _Float16 v16h;
typedef __attribute__((ext_vector_type(8)))  _Float16 v8h;
typedef __attribute__((ext_vector_type(8)))  float    v8f;

#define BDIM  2
#define TDIM  2048
#define DDIM  1024
#define HH    16
#define DH    64
#define NROWS (BDIM * TDIM)   // 4096

static __device__ __forceinline__ v8f wmma_f16(v16h a, v16h b, v8f c) {
    return __builtin_amdgcn_wmma_f32_16x16x32_f16(
        false, a, false, b, (short)0, c, false, false);
}

// 16-byte async global -> LDS copy (per-lane). Tracked by ASYNCcnt.
static __device__ __forceinline__ void async_cp16(void* lds_ptr, const void* gptr) {
    unsigned lds_off = (unsigned)(uintptr_t)lds_ptr;   // low 32 bits = LDS offset
    uint64_t gaddr   = (uint64_t)(uintptr_t)gptr;
    asm volatile("global_load_async_to_lds_b128 %0, %1, off"
                 :: "v"(lds_off), "v"(gaddr)
                 : "memory");
}
static __device__ __forceinline__ void wait_async0() {
    asm volatile("s_wait_asynccnt 0x0" ::: "memory");
}

// ---------------------------------------------------------------------------
// Kernel 1: LayerNorm, f32 in -> f16 out.  One block (256 thr) per row.
// ---------------------------------------------------------------------------
__global__ void ln_kernel(const float* __restrict__ x,
                          const float* __restrict__ gamma,
                          const float* __restrict__ beta,
                          _Float16* __restrict__ xn) {
    const int row = blockIdx.x;
    const int tid = threadIdx.x;
    const float* xr = x + (size_t)row * DDIM;
    float4 v = ((const float4*)xr)[tid];
    float s  = v.x + v.y + v.z + v.w;
    float s2 = v.x*v.x + v.y*v.y + v.z*v.z + v.w*v.w;
    #pragma unroll
    for (int m = 16; m >= 1; m >>= 1) {
        s  += __shfl_xor(s,  m, 32);
        s2 += __shfl_xor(s2, m, 32);
    }
    __shared__ float rs[8], rs2[8];
    const int wv = tid >> 5, ln = tid & 31;
    if (ln == 0) { rs[wv] = s; rs2[wv] = s2; }
    __syncthreads();
    if (tid == 0) {
        float a = 0.f, b = 0.f;
        #pragma unroll
        for (int i = 0; i < 8; i++) { a += rs[i]; b += rs2[i]; }
        rs[0] = a; rs2[0] = b;
    }
    __syncthreads();
    const float mu  = rs[0] * (1.0f / DDIM);
    const float var = rs2[0] * (1.0f / DDIM) - mu * mu;
    const float inv = rsqrtf(var + 1e-5f);
    const int base = tid * 4;
    float4 g  = ((const float4*)gamma)[tid];
    float4 bb = ((const float4*)beta)[tid];
    _Float16* dst = xn + (size_t)row * DDIM + base;
    dst[0] = (_Float16)((v.x - mu) * inv * g.x + bb.x);
    dst[1] = (_Float16)((v.y - mu) * inv * g.y + bb.y);
    dst[2] = (_Float16)((v.z - mu) * inv * g.z + bb.z);
    dst[3] = (_Float16)((v.w - mu) * inv * g.w + bb.w);
}

// ---------------------------------------------------------------------------
// Kernel 2: weights -> f16, TRANSPOSED (WT[n][k] = W[k][n]), packed 4x.
// Makes the GEMM B-tile a straight row-major copy (async-friendly).
// ---------------------------------------------------------------------------
__global__ void cvt_wT_kernel(const float* __restrict__ wq, const float* __restrict__ wk,
                              const float* __restrict__ wv, const float* __restrict__ wo,
                              _Float16* __restrict__ w16t) {
    const size_t i = ((size_t)blockIdx.x * blockDim.x + threadIdx.x) * 4;
    const size_t DD = (size_t)DDIM * DDIM;
    const float* srcs[4] = {wq, wk, wv, wo};
    const size_t m = i / DD, off = i % DD;
    const int n = (int)(off / DDIM);     // output row (= W column)
    const int k = (int)(off % DDIM);     // output col (= W row), 4 contiguous
    const float* s = srcs[m];
    _Float16* d = w16t + i;
    #pragma unroll
    for (int j = 0; j < 4; j++) d[j] = (_Float16)s[(size_t)(k + j) * DDIM + n];
}

// ---------------------------------------------------------------------------
// Kernel 3/5: tiled WMMA GEMM.  C[4096 x 1024] = A16 * WT16^T + bias.
//   Block 256 thr = 8 waves; tile 128(M) x 64(N); K-step 32.
//   B tile async-copied from WT (row-major [n][k]) into Bs[64][32].
// MODE 0: out f16 row-major (q, k).
// MODE 1: out f16 transposed [D][NROWS] (v -> VT, free in scattered epilogue).
// MODE 2: out f32 = val + resid (final output).
// ---------------------------------------------------------------------------
template <int MODE>
__global__ void gemm_kernel(const _Float16* __restrict__ A,
                            const _Float16* __restrict__ WT,
                            const float* __restrict__ bias,
                            const float* __restrict__ resid,
                            _Float16* __restrict__ out16,
                            float* __restrict__ out32) {
    __shared__ _Float16 Bs[64][32];    // [col][k], 4 KB
    const int tid = threadIdx.x;
    const int wv  = tid >> 5;
    const int ln  = tid & 31;
    const int M0  = blockIdx.y * 128 + wv * 16;
    const int N0  = blockIdx.x * 64;
    const int lrow  = ln & 15;
    const int lhalf = ln >> 4;

    v8f acc[4] = {v8f{}, v8f{}, v8f{}, v8f{}};
    const _Float16* arow = A + (size_t)(M0 + lrow) * DDIM;

    const int sc  = tid >> 2;          // 0..63 : Bs row (= output col)
    const int sk  = (tid & 3) * 8;     // 0..24 : k offset (16B chunks)
    const _Float16* wrow = WT + (size_t)(N0 + sc) * DDIM + sk;

    for (int k0 = 0; k0 < DDIM; k0 += 32) {
        async_cp16(&Bs[sc][sk], wrow + k0);
        if (k0 + 32 < DDIM)
            __builtin_prefetch(arow + k0 + 32 + lhalf * 16, 0, 0);
        wait_async0();
        __syncthreads();

        v16h afrag = *(const v16h*)(arow + k0 + lhalf * 16);
        #pragma unroll
        for (int j = 0; j < 4; j++) {
            v16h bfrag = *(const v16h*)&Bs[j * 16 + lrow][lhalf * 16];
            acc[j] = wmma_f16(afrag, bfrag, acc[j]);
        }
        __syncthreads();
    }

    const int rowbase = M0 + lhalf * 8;
    #pragma unroll
    for (int j = 0; j < 4; j++) {
        const int col = N0 + j * 16 + lrow;
        const float bv = bias[col];
        #pragma unroll
        for (int i = 0; i < 8; i++) {
            const int row = rowbase + i;
            const float val = acc[j][i] + bv;
            if (MODE == 2) {
                const size_t idx = (size_t)row * DDIM + col;
                out32[idx] = val + resid[idx];
            } else if (MODE == 1) {
                out16[(size_t)col * NROWS + row] = (_Float16)val;   // VT
            } else {
                out16[(size_t)row * DDIM + col] = (_Float16)val;
            }
        }
    }
}

// ---------------------------------------------------------------------------
// Kernel 4: flash-style attention per (b, h, 128-query-row block).
//   K tile, VT tile and mask tile all async-copied to LDS each m-step.
// ---------------------------------------------------------------------------
__global__ void attn_kernel(const _Float16* __restrict__ Q,
                            const _Float16* __restrict__ K,
                            const _Float16* __restrict__ VT,   // [D][NROWS]
                            const int* __restrict__ mask,
                            _Float16* __restrict__ Y) {
    __shared__ _Float16 Ks[32][64];      // [keyrow][k]       4 KB
    __shared__ _Float16 Vs[64][32];      // [d][keyrow]       4 KB
    __shared__ _Float16 Ps[8][16 * 32];  // per-wave P tiles  8 KB
    __shared__ int      Ms[128][32];     // mask tile        16 KB

    const int tid = threadIdx.x;
    const int wv  = tid >> 5;
    const int ln  = tid & 31;
    const int lrow  = ln & 15;
    const int lhalf = ln >> 4;
    const int b = blockIdx.z, h = blockIdx.y;
    const int nblk = blockIdx.x * 128;                // block's query-row base
    const int n0 = nblk + wv * 16;                    // this wave's rows
    const size_t headoff = (size_t)h * DH;

    const size_t qrow = (size_t)(b * TDIM + n0 + lrow) * DDIM + headoff;
    v16h q0 = *(const v16h*)(Q + qrow + lhalf * 16);
    v16h q1 = *(const v16h*)(Q + qrow + 32 + lhalf * 16);

    v8f o[4] = {v8f{}, v8f{}, v8f{}, v8f{}};
    float lsum[8] = {0.f, 0.f, 0.f, 0.f, 0.f, 0.f, 0.f, 0.f};
    const int lrow_c = wv * 16 + lhalf * 8;           // block-local C-layout row

    // staging thread maps
    const int kmr = tid >> 3, kc = (tid & 7) * 8;     // Ks: 32 x 64
    const int vd  = tid >> 2, vk = (tid & 3) * 8;     // Vs: 64 x 32
    const _Float16* ksrc = K  + (size_t)(b * TDIM + kmr) * DDIM + headoff + kc;
    const _Float16* vsrc = VT + (headoff + vd) * (size_t)NROWS + b * TDIM + vk;

    for (int m0 = 0; m0 < TDIM; m0 += 32) {
        async_cp16(&Ks[kmr][kc], ksrc + (size_t)m0 * DDIM);
        async_cp16(&Vs[vd][vk],  vsrc + m0);
        #pragma unroll
        for (int u = 0; u < 4; u++) {                 // mask tile 128x32 ints
            const int c   = tid + u * 256;            // 16B chunk id, 0..1023
            const int mrw = c >> 3;
            const int mcl = (c & 7) * 4;
            async_cp16(&Ms[mrw][mcl],
                       mask + ((size_t)b * TDIM + nblk + mrw) * TDIM + m0 + mcl);
        }
        wait_async0();
        __syncthreads();

        v8f s[2];
        #pragma unroll
        for (int t = 0; t < 2; t++) {
            v16h kf0 = *(const v16h*)&Ks[t * 16 + lrow][lhalf * 16];
            v16h kf1 = *(const v16h*)&Ks[t * 16 + lrow][32 + lhalf * 16];
            v8f z = v8f{};
            z = wmma_f16(q0, kf0, z);
            z = wmma_f16(q1, kf1, z);
            s[t] = z;
        }

        #pragma unroll
        for (int t = 0; t < 2; t++) {
            const int mloc = t * 16 + lrow;
            #pragma unroll
            for (int i = 0; i < 8; i++) {
                const int mk = Ms[lrow_c + i][mloc];
                float p = mk ? __expf(s[t][i] * 0.125f) : 0.0f;
                float r = p;
                r += __shfl_xor(r, 1, 32);
                r += __shfl_xor(r, 2, 32);
                r += __shfl_xor(r, 4, 32);
                r += __shfl_xor(r, 8, 32);
                lsum[i] += r;
                Ps[wv][(i + lhalf * 8) * 32 + mloc] = (_Float16)p;
            }
        }

        v16h pf = *(const v16h*)&Ps[wv][lrow * 32 + lhalf * 16];
        #pragma unroll
        for (int j = 0; j < 4; j++) {
            v16h vf = *(const v16h*)&Vs[j * 16 + lrow][lhalf * 16];
            o[j] = wmma_f16(pf, vf, o[j]);
        }
        __syncthreads();
    }

    #pragma unroll
    for (int j = 0; j < 4; j++) {
        #pragma unroll
        for (int i = 0; i < 8; i++) {
            const int nrow = n0 + lhalf * 8 + i;
            const float val = o[j][i] / (lsum[i] + 1e-20f);
            Y[(size_t)(b * TDIM + nrow) * DDIM + headoff + j * 16 + lrow] = (_Float16)val;
        }
    }
}

// ---------------------------------------------------------------------------
// Host launcher
// ---------------------------------------------------------------------------
extern "C" void kernel_launch(void* const* d_in, const int* in_sizes, int n_in,
                              void* d_out, int out_size, void* d_ws, size_t ws_size,
                              hipStream_t stream) {
    const float* x     = (const float*)d_in[0];
    const int*   mask  = (const int*)  d_in[1];
    const float* gamma = (const float*)d_in[2];
    const float* beta  = (const float*)d_in[3];
    const float* Wq    = (const float*)d_in[4];
    const float* bq    = (const float*)d_in[5];
    const float* Wk    = (const float*)d_in[6];
    const float* bk    = (const float*)d_in[7];
    const float* Wv    = (const float*)d_in[8];
    const float* bv    = (const float*)d_in[9];
    const float* Wo    = (const float*)d_in[10];
    const float* bo    = (const float*)d_in[11];
    float* out = (float*)d_out;

    const size_t ND = (size_t)NROWS * DDIM;
    const size_t DD = (size_t)DDIM * DDIM;
    _Float16* xn   = (_Float16*)d_ws;
    _Float16* q    = xn + ND;
    _Float16* k    = q + ND;
    _Float16* vt   = k + ND;          // [D][NROWS]
    _Float16* y    = vt + ND;
    _Float16* w16t = y + ND;          // 4*DD halves (total 48 MB)

    ln_kernel<<<NROWS, 256, 0, stream>>>(x, gamma, beta, xn);
    cvt_wT_kernel<<<(4 * DD) / 1024, 256, 0, stream>>>(Wq, Wk, Wv, Wo, w16t);

    dim3 ggrid(DDIM / 64, NROWS / 128);
    gemm_kernel<0><<<ggrid, 256, 0, stream>>>(xn, w16t + 0 * DD, bq, nullptr, q, nullptr);
    gemm_kernel<0><<<ggrid, 256, 0, stream>>>(xn, w16t + 1 * DD, bk, nullptr, k, nullptr);
    gemm_kernel<1><<<ggrid, 256, 0, stream>>>(xn, w16t + 2 * DD, bv, nullptr, vt, nullptr);

    dim3 agrid(TDIM / 128, HH, BDIM);
    attn_kernel<<<agrid, 256, 0, stream>>>(q, k, vt, mask, y);

    gemm_kernel<2><<<ggrid, 256, 0, stream>>>(y, w16t + 3 * DD, bo, x, nullptr, out);
}
